// get_model_57801669869946
// MI455X (gfx1250) — matrix operand order repaired and checked by reading, not compiled, not run on hardware
//
#include <hip/hip_runtime.h>
#include <hip/hip_bf16.h>

typedef __attribute__((ext_vector_type(16))) _Float16 v16h;
typedef __attribute__((ext_vector_type(8)))  _Float16 v8h;
typedef __attribute__((ext_vector_type(8)))  float    v8f;

#define BN_SCALE 0.99999500003749968f   /* 1/sqrt(1+1e-5) */

// -------------------- transpose input [B,9,N] -> [B,N,9] --------------------
__global__ __launch_bounds__(256) void k_transpose_in(const float* __restrict__ in,
                                                      float* __restrict__ out,
                                                      int B, int N) {
  int tid = blockIdx.x * 256 + threadIdx.x;
  if (tid >= B * N) return;
  int b = tid / N, n = tid % N;
  #pragma unroll
  for (int c = 0; c < 9; ++c)
    out[((size_t)b * N + n) * 9 + c] = in[((size_t)b * 9 + c) * N + n];
}

// -------------------- farthest point sampling (1 block per batch) ----------
__global__ __launch_bounds__(256) void k_fps(const float* __restrict__ coords, int ld,
                                             int N, int S, float* __restrict__ outc) {
  __shared__ float dist[4096];
  __shared__ float redv[256];
  __shared__ int   redi[256];
  int b = blockIdx.x;
  int t = threadIdx.x;
  const float* pc = coords + (size_t)b * N * ld;
  float* oc = outc + (size_t)b * S * 3;
  for (int n = t; n < N; n += 256) dist[n] = 1e10f;
  __syncthreads();
  int far = 0;
  for (int it = 0; it < S; ++it) {
    float fx = pc[(size_t)far * ld + 0];
    float fy = pc[(size_t)far * ld + 1];
    float fz = pc[(size_t)far * ld + 2];
    if (t == 0) { oc[it * 3 + 0] = fx; oc[it * 3 + 1] = fy; oc[it * 3 + 2] = fz; }
    float best = -1.0f; int bi = 0;
    for (int n = t; n < N; n += 256) {
      float dx = pc[(size_t)n * ld + 0] - fx;
      float dy = pc[(size_t)n * ld + 1] - fy;
      float dz = pc[(size_t)n * ld + 2] - fz;
      float d  = dx * dx + dy * dy + dz * dz;
      float dm = fminf(dist[n], d);
      dist[n] = dm;
      if (dm > best) { best = dm; bi = n; }
    }
    redv[t] = best; redi[t] = bi;
    __syncthreads();
    for (int s2 = 128; s2 > 0; s2 >>= 1) {
      if (t < s2) {
        float ov = redv[t + s2]; int oi = redi[t + s2];
        if (ov > redv[t] || (ov == redv[t] && oi < redi[t])) { redv[t] = ov; redi[t] = oi; }
      }
      __syncthreads();
    }
    far = redi[0];
    __syncthreads();
  }
}

// -------------------- ball query: first <=32 hits in index order -----------
__global__ __launch_bounds__(256) void k_ball_query(const float* __restrict__ centers,
                                                    const float* __restrict__ coords, int ldc,
                                                    int N, int S, float r2, int total,
                                                    int* __restrict__ gidx) {
  int tid = blockIdx.x * 256 + threadIdx.x;
  if (tid >= total) return;
  int b = tid / S;
  const float* ct = centers + (size_t)tid * 3;
  float cx = ct[0], cy = ct[1], cz = ct[2];
  const float* pc = coords + (size_t)b * N * ldc;
  int* o = gidx + (size_t)tid * 32;
  int cnt = 0;
  for (int n = 0; n < N && cnt < 32; ++n) {
    float dx = pc[(size_t)n * ldc + 0] - cx;
    float dy = pc[(size_t)n * ldc + 1] - cy;
    float dz = pc[(size_t)n * ldc + 2] - cz;
    if (dx * dx + dy * dy + dz * dz <= r2) o[cnt++] = n;
  }
  if (cnt == 0) o[cnt++] = 0;
  int f0 = o[0];
  for (; cnt < 32; ++cnt) o[cnt] = f0;
}

// -------------------- build grouped features (f16, K-padded) ---------------
__global__ __launch_bounds__(256) void k_group(const int* __restrict__ gidx,
                                               const float* __restrict__ coords, int ldc,
                                               const float* __restrict__ centers,
                                               const float* __restrict__ feats, int D,
                                               int N, int S, int Kpad, int total,
                                               _Float16* __restrict__ out) {
  int tid = blockIdx.x * 256 + threadIdx.x;
  if (tid >= total) return;
  int bs = tid >> 5;
  int b = bs / S;
  int gi = gidx[tid];
  const float* pc = coords + ((size_t)b * N + gi) * ldc;
  const float* ct = centers + (size_t)bs * 3;
  _Float16* o = out + (size_t)tid * Kpad;
  o[0] = (_Float16)(pc[0] - ct[0]);
  o[1] = (_Float16)(pc[1] - ct[1]);
  o[2] = (_Float16)(pc[2] - ct[2]);
  const float* f = feats + ((size_t)b * N + gi) * (size_t)D;
  for (int j = 0; j < D; ++j) o[3 + j] = (_Float16)f[j];
  for (int c = 3 + D; c < Kpad; ++c) o[c] = (_Float16)0.0f;
}

// ---- fold BN into weights; f32 -> f16, K-padded, WMMA-B-fragment swizzled --
// Packed layout: element for (ktile kt, ntile nt, lane l, slot j) lives at
//   ((kt*ntiles + nt)*32 + l)*16 + j
// which maps to logical W[k][c] with k = kt*32 + ((l>=16)?16:0) + j,
//                                  c = nt*16 + (l&15)
// so each lane's 16 B-fragment halves are 32 contiguous, 32B-aligned bytes.
__global__ __launch_bounds__(256) void k_prep_weights(const float* __restrict__ W,
                                                      const float* __restrict__ b,
                                                      const float* __restrict__ g,
                                                      const float* __restrict__ bt,
                                                      _Float16* __restrict__ W16,
                                                      float* __restrict__ bias,
                                                      int Cin, int Kpad, int Cout) {
  int tid = blockIdx.x * 256 + threadIdx.x;
  if (tid >= Kpad * Cout) return;
  int j    = tid & 15;
  int l    = (tid >> 4) & 31;
  int tile = tid >> 9;
  int ntiles = Cout >> 4;
  int nt = tile % ntiles;
  int kt = tile / ntiles;
  int k = (kt << 5) + ((l >> 4) << 4) + j;
  int c = (nt << 4) + (l & 15);
  float s = g[c] * BN_SCALE;
  float v = (k < Cin) ? W[(size_t)k * Cout + c] * s : 0.0f;
  W16[tid] = (_Float16)v;
  if (k == 0) bias[c] = b[c] * s + bt[c];
}

// -------------------- WMMA GEMM + bias + ReLU ------------------------------
// X:   f16 [M,K] row-major (K % 32 == 0, M % 16 == 0, pad cols zeroed)
// Wpk: f16 packed B-fragments (see k_prep_weights), bias f32 [N]
// Y:   f16 [M,N]; Y32 optional f32 copy.
// One wave owns a 16x32 tile of Y (1 M-tile x 2 N-tiles); 4 waves per block.
// grid = (ceil(M/16/4), N/32). N % 32 == 0 for every layer in this model.
__global__ __launch_bounds__(128) void k_gemm_bn_relu(const _Float16* __restrict__ X,
                                                      const _Float16* __restrict__ Wpk,
                                                      const float* __restrict__ bias,
                                                      _Float16* __restrict__ Y,
                                                      float* __restrict__ Y32,
                                                      int M, int K, int N) {
  int wave = threadIdx.x >> 5;
  int lane = threadIdx.x & 31;
  int tm = blockIdx.x * 4 + wave;          // wave-uniform: EXEC all-1s at WMMA
  if (tm * 16 >= M) return;
  int ntiles = N >> 4;
  int nt0 = blockIdx.y << 1;               // two adjacent n-tiles

  int arow  = (tm << 4) + (lane & 15);     // A: lane = row, K-half by lane group
  int akoff = (lane >> 4) << 4;            // 0 or 16
  const _Float16* xrow  = X + (size_t)arow * K + akoff;
  const _Float16* wbase = Wpk + (((size_t)nt0 * 32) + lane) * 16;  // kt = 0
  const size_t wstride  = (size_t)ntiles * 512;                    // halves per ktile

  v8f acc0 = {}, acc1 = {};
  for (int k0 = 0; k0 < K; k0 += 32) {
    __builtin_prefetch(xrow + k0 + 128, 0, 1);   // global_prefetch_b8, speculative
    v8h lo = *(const v8h*)(xrow + k0);           // K = k0 + akoff + 0..7
    v8h hi = *(const v8h*)(xrow + k0 + 8);       // K = k0 + 8 + akoff + 0..7
    v16h a = __builtin_shufflevector(lo, hi, 0, 1, 2, 3, 4, 5, 6, 7,
                                             8, 9, 10, 11, 12, 13, 14, 15);
    const _Float16* wp = wbase + (size_t)(k0 >> 5) * wstride;
    v16h b0 = *(const v16h*)(wp);                // 32B contiguous per lane
    v16h b1 = *(const v16h*)(wp + 512);          // next n-tile's fragments
    acc0 = __builtin_amdgcn_wmma_f32_16x16x32_f16(false, a, false, b0,
                                                  (short)0, acc0, false, false);
    acc1 = __builtin_amdgcn_wmma_f32_16x16x32_f16(false, a, false, b1,
                                                  (short)0, acc1, false, false);
  }
  // C/D layout: VGPR r -> row (tm*16 + r + 8*(lane>=16)), col = nt*16 + lane&15
  int rbase = (tm << 4) + ((lane >> 4) << 3);
  int c0 = (nt0 << 4) + (lane & 15);
  float bc0 = bias[c0];
  float bc1 = bias[c0 + 16];
  #pragma unroll
  for (int r = 0; r < 8; ++r) {
    int rr = rbase + r;
    float v0 = fmaxf(acc0[r] + bc0, 0.0f);
    float v1 = fmaxf(acc1[r] + bc1, 0.0f);
    Y[(size_t)rr * N + c0]      = (_Float16)v0;
    Y[(size_t)rr * N + c0 + 16] = (_Float16)v1;
    if (Y32) {
      Y32[(size_t)rr * N + c0]      = v0;
      Y32[(size_t)rr * N + c0 + 16] = v1;
    }
  }
}

// -------------------- max over 32 grouped samples --------------------------
__global__ __launch_bounds__(256) void k_maxpool32(const _Float16* __restrict__ Y, int C,
                                                   int total, float* __restrict__ F) {
  int tid = blockIdx.x * 256 + threadIdx.x;
  if (tid >= total) return;
  int bs = tid / C, c = tid % C;
  const _Float16* y = Y + (size_t)bs * 32 * C + c;
  float m = -3.0e38f;
  #pragma unroll 4
  for (int k = 0; k < 32; ++k) m = fmaxf(m, (float)y[(size_t)k * C]);
  F[tid] = m;
}

// -------------------- 3-NN inverse-distance interpolation + concat ---------
__global__ __launch_bounds__(256) void k_knn_interp(const float* __restrict__ x1, int ld1, int N1,
                                                    const float* __restrict__ x2, int ld2, int N2,
                                                    const float* __restrict__ P1, int D1,
                                                    const float* __restrict__ P2, int D2,
                                                    int Kpad, int total,
                                                    _Float16* __restrict__ out) {
  int tid = blockIdx.x * 256 + threadIdx.x;
  if (tid >= total) return;
  int b = tid / N1, n = tid % N1;
  const float* a = x1 + ((size_t)b * N1 + n) * ld1;
  float ax = a[0], ay = a[1], az = a[2];
  const float* q = x2 + (size_t)b * N2 * ld2;
  float bd0 = 3.0e38f, bd1 = 3.0e38f, bd2 = 3.0e38f;
  int bi0 = 0, bi1 = 0, bi2 = 0;
  for (int m = 0; m < N2; ++m) {
    float dx = ax - q[(size_t)m * ld2 + 0];
    float dy = ay - q[(size_t)m * ld2 + 1];
    float dz = az - q[(size_t)m * ld2 + 2];
    float d = dx * dx + dy * dy + dz * dz;
    if      (d < bd0) { bd2 = bd1; bi2 = bi1; bd1 = bd0; bi1 = bi0; bd0 = d; bi0 = m; }
    else if (d < bd1) { bd2 = bd1; bi2 = bi1; bd1 = d; bi1 = m; }
    else if (d < bd2) { bd2 = d; bi2 = m; }
  }
  float w0 = 1.0f / (bd0 + 1e-8f);
  float w1 = 1.0f / (bd1 + 1e-8f);
  float w2 = 1.0f / (bd2 + 1e-8f);
  float ws = w0 + w1 + w2; w0 /= ws; w1 /= ws; w2 /= ws;
  _Float16* o = out + (size_t)tid * Kpad;
  int c = 0;
  if (P1) {
    const float* p1 = P1 + ((size_t)b * N1 + n) * D1;
    for (int j = 0; j < D1; ++j) o[c++] = (_Float16)p1[j];
  }
  const float* p20 = P2 + ((size_t)b * N2 + bi0) * D2;
  const float* p21 = P2 + ((size_t)b * N2 + bi1) * D2;
  const float* p22 = P2 + ((size_t)b * N2 + bi2) * D2;
  for (int j = 0; j < D2; ++j)
    o[c++] = (_Float16)(w0 * p20[j] + w1 * p21[j] + w2 * p22[j]);
  for (; c < Kpad; ++c) o[c] = (_Float16)0.0f;
}

// -------------------- head: h(64) @ W2(64x13) + b2, softmax ----------------
__global__ __launch_bounds__(256) void k_head_out(const _Float16* __restrict__ H,
                                                  const float* __restrict__ W2,
                                                  const float* __restrict__ b2,
                                                  float* __restrict__ logits,
                                                  float* __restrict__ probs, int total) {
  int tid = blockIdx.x * 256 + threadIdx.x;
  if (tid >= total) return;
  const _Float16* h = H + (size_t)tid * 64;
  float hv[64];
  #pragma unroll
  for (int k = 0; k < 64; ++k) hv[k] = (float)h[k];
  float lg[13];
  #pragma unroll
  for (int c = 0; c < 13; ++c) {
    float s = b2[c];
    for (int k = 0; k < 64; ++k) s += hv[k] * W2[(size_t)k * 13 + c];
    lg[c] = s;
  }
  float m = lg[0];
  for (int c = 1; c < 13; ++c) m = fmaxf(m, lg[c]);
  float e[13], sum = 0.0f;
  for (int c = 0; c < 13; ++c) { e[c] = __expf(lg[c] - m); sum += e[c]; }
  float inv = 1.0f / sum;
  float* lo = logits + (size_t)tid * 13;
  float* po = probs  + (size_t)tid * 13;
  for (int c = 0; c < 13; ++c) { lo[c] = lg[c]; po[c] = e[c] * inv; }
}

// -------------------- l4_points [B,16,512] -> [B,512,16] -------------------
__global__ __launch_bounds__(256) void k_emit_l4(const float* __restrict__ F4,
                                                 float* __restrict__ out, int B) {
  int tid = blockIdx.x * 256 + threadIdx.x;
  if (tid >= B * 512 * 16) return;
  int b = tid / (512 * 16);
  int r = tid % (512 * 16);
  int c = r / 16, s = r % 16;
  out[tid] = F4[((size_t)b * 16 + s) * 512 + c];
}

// ============================================================================
extern "C" void kernel_launch(void* const* d_in, const int* in_sizes, int n_in,
                              void* d_out, int out_size, void* d_ws, size_t ws_size,
                              hipStream_t stream) {
  (void)in_sizes; (void)n_in; (void)out_size; (void)ws_size;
  const int B = 8, N0 = 4096;
  const float* xyz = (const float*)d_in[0];

  char* base = (char*)d_ws;
  size_t off = 0;
  auto alloc = [&](size_t bytes) -> void* {
    off = (off + 255) & ~(size_t)255;
    void* p = base + off;
    off += bytes;
    return p;
  };

  float* P0 = (float*)alloc((size_t)B * N0 * 9 * 4);      // transposed input / ps0
  float* C1 = (float*)alloc((size_t)B * 1024 * 3 * 4);
  float* C2 = (float*)alloc((size_t)B * 256 * 3 * 4);
  float* C3 = (float*)alloc((size_t)B * 64 * 3 * 4);
  float* C4 = (float*)alloc((size_t)B * 16 * 3 * 4);
  float* F1 = (float*)alloc((size_t)B * 1024 * 64 * 4);   // SA outputs (f32)
  float* F2 = (float*)alloc((size_t)B * 256 * 128 * 4);
  float* F3 = (float*)alloc((size_t)B * 64 * 256 * 4);
  float* F4 = (float*)alloc((size_t)B * 16 * 512 * 4);
  float* Q3 = (float*)alloc((size_t)B * 64 * 256 * 4);    // FP outputs (f32)
  float* Q2 = (float*)alloc((size_t)B * 256 * 128 * 4);
  float* Q1 = (float*)alloc((size_t)B * 1024 * 64 * 4);
  int* gidx = (int*)alloc((size_t)B * 1024 * 32 * 4);
  const size_t actEl = (size_t)B * 1024 * 32 * 64;        // max rows(262144) * max cols(64)
  _Float16* actA = (_Float16*)alloc(actEl * 2);
  _Float16* actB = (_Float16*)alloc(actEl * 2);

  // 22 GEMM layers: d_in W index, Cin, padded K, Cout
  static const int Lw[22]   = {1,5,9, 13,17,21, 25,29,33, 37,41,45, 49,53, 57,61, 65,69, 73,77,81, 85};
  static const int Lcin[22] = {12,32,32, 67,64,64, 131,128,128, 259,256,256, 768,256, 384,256, 192,128, 64,64,64, 64};
  static const int Lk[22]   = {32,32,32, 96,64,64, 160,128,128, 288,256,256, 768,256, 384,256, 192,128, 64,64,64, 64};
  static const int Lco[22]  = {32,32,64, 64,64,128, 128,128,256, 256,256,512, 256,256, 256,128, 128,64, 64,64,64, 64};

  _Float16* Wp[22]; float* Bp[22];
  for (int i = 0; i < 22; ++i) {
    Wp[i] = (_Float16*)alloc((size_t)Lk[i] * Lco[i] * 2);
    Bp[i] = (float*)alloc((size_t)Lco[i] * 4);
    int tot = Lk[i] * Lco[i];
    k_prep_weights<<<(tot + 255) / 256, 256, 0, stream>>>(
        (const float*)d_in[Lw[i]], (const float*)d_in[Lw[i] + 1],
        (const float*)d_in[Lw[i] + 2], (const float*)d_in[Lw[i] + 3],
        Wp[i], Bp[i], Lcin[i], Lk[i], Lco[i]);
  }

  auto gemm = [&](const _Float16* X, int li, _Float16* Y, float* Y32, int M) {
    dim3 g((M / 16 + 3) / 4, Lco[li] / 32);   // 2 n-tiles per wave
    k_gemm_bn_relu<<<g, 128, 0, stream>>>(X, Wp[li], Bp[li], Y, Y32, M, Lk[li], Lco[li]);
  };

  k_transpose_in<<<(B * N0 + 255) / 256, 256, 0, stream>>>(xyz, P0, B, N0);

  // ---------------- Set Abstraction stages ----------------
  const int   saN[4] = {4096, 1024, 256, 64};
  const int   saS[4] = {1024, 256, 64, 16};
  const float saR[4] = {0.1f, 0.2f, 0.4f, 0.8f};
  const int   saD[4] = {9, 64, 128, 256};
  const int   saK[4] = {32, 96, 160, 288};
  const float* cinp[4] = {P0, C1, C2, C3};
  const int    ldcs[4] = {9, 3, 3, 3};
  float*       coutp[4] = {C1, C2, C3, C4};
  const float* finp[4] = {P0, F1, F2, F3};
  float*       foutp[4] = {F1, F2, F3, F4};
  const int    lb[4] = {0, 3, 6, 9};

  for (int s = 0; s < 4; ++s) {
    k_fps<<<B, 256, 0, stream>>>(cinp[s], ldcs[s], saN[s], saS[s], coutp[s]);
    int BS = B * saS[s];
    k_ball_query<<<(BS + 255) / 256, 256, 0, stream>>>(
        coutp[s], cinp[s], ldcs[s], saN[s], saS[s], saR[s] * saR[s], BS, gidx);
    int rows = BS * 32;
    k_group<<<(rows + 255) / 256, 256, 0, stream>>>(
        gidx, cinp[s], ldcs[s], coutp[s], finp[s], saD[s], saN[s], saS[s], saK[s], rows, actA);
    _Float16* X = actA; _Float16* Y = actB;
    for (int l = 0; l < 3; ++l) {
      gemm(X, lb[s] + l, Y, nullptr, rows);
      _Float16* t = X; X = Y; Y = t;     // X now holds latest output
    }
    int Cf = Lco[lb[s] + 2];
    k_maxpool32<<<(BS * Cf + 255) / 256, 256, 0, stream>>>(X, Cf, BS * Cf, foutp[s]);
  }

  // ---------------- Feature Propagation stages ----------------
  { // FP0: l3 <- l4
    int BN1 = B * 64;
    k_knn_interp<<<(BN1 + 255) / 256, 256, 0, stream>>>(
        C3, 3, 64, C4, 3, 16, F3, 256, F4, 512, 768, BN1, actA);
    _Float16 *X = actA, *Y = actB;
    gemm(X, 12, Y, nullptr, BN1); { _Float16* t = X; X = Y; Y = t; }
    gemm(X, 13, Y, Q3, BN1);
  }
  { // FP1: l2 <- l3
    int BN1 = B * 256;
    k_knn_interp<<<(BN1 + 255) / 256, 256, 0, stream>>>(
        C2, 3, 256, C3, 3, 64, F2, 128, Q3, 256, 384, BN1, actA);
    _Float16 *X = actA, *Y = actB;
    gemm(X, 14, Y, nullptr, BN1); { _Float16* t = X; X = Y; Y = t; }
    gemm(X, 15, Y, Q2, BN1);
  }
  { // FP2: l1 <- l2
    int BN1 = B * 1024;
    k_knn_interp<<<(BN1 + 255) / 256, 256, 0, stream>>>(
        C1, 3, 1024, C2, 3, 256, F1, 64, Q2, 128, 192, BN1, actA);
    _Float16 *X = actA, *Y = actB;
    gemm(X, 16, Y, nullptr, BN1); { _Float16* t = X; X = Y; Y = t; }
    gemm(X, 17, Y, Q1, BN1);
  }
  { // FP3: l0 <- l1, then head
    int BN1 = B * 4096;
    k_knn_interp<<<(BN1 + 255) / 256, 256, 0, stream>>>(
        P0, 9, 4096, C1, 3, 1024, nullptr, 0, Q1, 64, 64, BN1, actA);
    _Float16 *X = actA, *Y = actB;
    gemm(X, 18, Y, nullptr, BN1); { _Float16* t = X; X = Y; Y = t; }
    gemm(X, 19, Y, nullptr, BN1); { _Float16* t = X; X = Y; Y = t; }
    gemm(X, 20, Y, nullptr, BN1); { _Float16* t = X; X = Y; Y = t; }   // p0
    gemm(X, 21, Y, nullptr, BN1); { _Float16* t = X; X = Y; Y = t; }   // head FC1 -> h

    float* outp   = (float*)d_out;
    float* logits = outp;
    float* l4     = outp + (size_t)B * 4096 * 13;
    float* probs  = l4 + (size_t)B * 512 * 16;
    k_head_out<<<(BN1 + 255) / 256, 256, 0, stream>>>(
        X, (const float*)d_in[89], (const float*)d_in[90], logits, probs, BN1);
    k_emit_l4<<<(B * 512 * 16 + 255) / 256, 256, 0, stream>>>(F4, l4, B);
  }
}
